// Transformer_P_90486370992257
// MI455X (gfx1250) — matrix operand order repaired and checked by reading, not compiled, not run on hardware
//
#include <hip/hip_runtime.h>

#define BATCH   2
#define NFEAT   6000
#define DMODEL  512
#define HEADS   8
#define DH      64
#define LM      256
#define NTOK    6085      /* 1 + 78*78 */
#define NPAD    6144
#define PADF    59        /* NPAD - NTOK, zeros padded at FRONT */
#define LSEG    24        /* NPAD / LM */
#define D3      1536
#define BH      16        /* BATCH*HEADS */
#define HS      78
#define RESK    33
#define SCALEQ  0.125f    /* 64^-0.5 */

#define LDSPAD  40        /* padded inner stride (ushorts): 16B-aligned, bank-walked */

typedef __attribute__((ext_vector_type(16))) __bf16 v16bf;
typedef __attribute__((ext_vector_type(8)))  float  v8f;
typedef __attribute__((ext_vector_type(2)))  float  v2f;

union FragBF { v16bf v; unsigned short u[16]; unsigned int u32[8]; };

__device__ __forceinline__ unsigned short f2bf(float f) {
  union { float f; unsigned int u; } c; c.f = f;
  unsigned int r = c.u + 0x7FFFu + ((c.u >> 16) & 1u);
  return (unsigned short)(r >> 16);
}
__device__ __forceinline__ float bf2f(unsigned short h) {
  union { unsigned int u; float f; } c; c.u = ((unsigned int)h) << 16;
  return c.f;
}

// ---------------------------------------------------------------------------
// fp32 -> bf16 convert
// ---------------------------------------------------------------------------
__global__ void convert_bf16_k(const float* __restrict__ in,
                               unsigned short* __restrict__ out, long n) {
  long i = (long)blockIdx.x * blockDim.x + threadIdx.x;
  if (i < n) out[i] = f2bf(in[i]);
}

// ---------------------------------------------------------------------------
// Build h = concat([cls, features, features[:, :84]]) : (2, 6085, 512)
// ---------------------------------------------------------------------------
__global__ void build_h_k(const float* __restrict__ features,
                          const float* __restrict__ cls,
                          float* __restrict__ h) {
  long idx = (long)blockIdx.x * blockDim.x + threadIdx.x;
  long total = (long)BATCH * NTOK * DMODEL;
  if (idx >= total) return;
  int c = (int)(idx % DMODEL);
  long row = idx / DMODEL;
  int b = (int)(row / NTOK);
  int r = (int)(row % NTOK);
  if (r == 0) { h[idx] = cls[c]; return; }
  int i = r - 1;
  int src = (i < NFEAT) ? i : (i - NFEAT);
  h[idx] = features[((long)b * NFEAT + src) * DMODEL + c];
}

// ---------------------------------------------------------------------------
// LayerNorm + front-pad with zeros, writes bf16 x_pad (2, 6144, 512)
// ---------------------------------------------------------------------------
__global__ void __launch_bounds__(256)
ln_pad_k(const float* __restrict__ h, const float* __restrict__ g,
         const float* __restrict__ be, unsigned short* __restrict__ xp) {
  int j = blockIdx.x, b = blockIdx.y, t = threadIdx.x;
  unsigned short* orow = xp + ((long)b * NPAD + j) * DMODEL;
  if (j < PADF) { orow[t] = 0; orow[t + 256] = 0; return; }
  const float* x = h + ((long)b * NTOK + (j - PADF)) * DMODEL;
  float v0 = x[t], v1 = x[t + 256];
  __shared__ float red[256];
  red[t] = v0 + v1; __syncthreads();
  for (int s = 128; s > 0; s >>= 1) { if (t < s) red[t] += red[t + s]; __syncthreads(); }
  float mu = red[0] * (1.0f / DMODEL); __syncthreads();
  float d0 = v0 - mu, d1 = v1 - mu;
  red[t] = d0 * d0 + d1 * d1; __syncthreads();
  for (int s = 128; s > 0; s >>= 1) { if (t < s) red[t] += red[t + s]; __syncthreads(); }
  float rstd = rsqrtf(red[0] * (1.0f / DMODEL) + 1e-5f);
  orow[t]       = f2bf(d0 * rstd * g[t]       + be[t]);
  orow[t + 256] = f2bf(d1 * rstd * g[t + 256] + be[t + 256]);
}

// ---------------------------------------------------------------------------
// Final LayerNorm -> fp32 output
// ---------------------------------------------------------------------------
__global__ void __launch_bounds__(256)
ln_out_k(const float* __restrict__ h, const float* __restrict__ g,
         const float* __restrict__ be, float* __restrict__ out) {
  int j = blockIdx.x, b = blockIdx.y, t = threadIdx.x;
  const float* x = h + ((long)b * NTOK + j) * DMODEL;
  float* o = out + ((long)b * NTOK + j) * DMODEL;
  float v0 = x[t], v1 = x[t + 256];
  __shared__ float red[256];
  red[t] = v0 + v1; __syncthreads();
  for (int s = 128; s > 0; s >>= 1) { if (t < s) red[t] += red[t + s]; __syncthreads(); }
  float mu = red[0] * (1.0f / DMODEL); __syncthreads();
  float d0 = v0 - mu, d1 = v1 - mu;
  red[t] = d0 * d0 + d1 * d1; __syncthreads();
  for (int s = 128; s > 0; s >>= 1) { if (t < s) red[t] += red[t + s]; __syncthreads(); }
  float rstd = rsqrtf(red[0] * (1.0f / DMODEL) + 1e-5f);
  o[t]       = d0 * rstd * g[t]       + be[t];
  o[t + 256] = d1 * rstd * g[t + 256] + be[t + 256];
}

// ---------------------------------------------------------------------------
// bf16 WMMA GEMM:  C(M,N) = A(M,K) * B(K,N)  [optionally B given as Bt(N,K)]
//   - 256 threads = 8 waves; block tile 128x64; wave grid 4x2, each wave a
//     32x32 patch -> 4 v_wmma_f32_16x16x32_bf16 per 32-K step.
//   - LDS A row-major As[row][k]; LDS B COLUMN-major Bs[n][k] so every
//     fragment (A and B) is two contiguous ds_load_b128 per ISA 7.12.2.
//   - Cf : fp32 output (ldc, sC)  (nullable)
//   - Cb : bf16 output; headMode=1 writes (bz, n>>6, m, n&63) head layout
//   - bias added per output column, outScale applied to accum
// ---------------------------------------------------------------------------
__global__ void __launch_bounds__(256)
gemm_bf16_k(const unsigned short* __restrict__ A,
            const unsigned short* __restrict__ B,
            float* __restrict__ Cf, unsigned short* __restrict__ Cb,
            const float* __restrict__ bias,
            int M, int N, int K, int lda, int ldb, int ldc,
            long sA, long sB, long sC,
            int bTrans, int headMode, float outScale) {
  __shared__ unsigned short As[128][LDSPAD];  // 128x32 tile, padded stride
  __shared__ unsigned short Bs[64][LDSPAD];   // column-major: Bs[n][k]
  int bz = blockIdx.z;
  const unsigned short* Ab = A + (long)bz * sA;
  const unsigned short* Bb = B + (long)bz * sB;
  int tm = blockIdx.x * 128;
  int tn = blockIdx.y * 64;
  int tid  = threadIdx.x;
  int wv   = tid >> 5;
  int lane = tid & 31;
  int m_off = (wv >> 1) * 32;   // 0,32,64,96
  int n_off = (wv & 1) * 32;    // 0,32
  int r  = lane & 15;
  int hi = lane >> 4;
  v8f acc00 = {}, acc01 = {}, acc10 = {}, acc11 = {};

  // per-thread global-load coordinates
  int arow = tid >> 1, acol = (tid & 1) * 16;     // A: 128x32, 16 elems/thread
  int bkr  = tid >> 3, bnc  = (tid & 7) * 8;      // B row-major path
  int btn  = tid >> 2, btk  = (tid & 3) * 8;      // B transposed path

  for (int k0 = 0; k0 < K; k0 += 32) {
    __syncthreads();
    { // A tile 128x32 (two b128 per thread)
      const unsigned short* src = Ab + (long)(tm + arow) * lda + k0 + acol;
      *(uint4*)(&As[arow][acol])     = *(const uint4*)(src);
      *(uint4*)(&As[arow][acol + 8]) = *(const uint4*)(src + 8);
    }
    if (!bTrans) { // B row-major KxN: coalesced b128 load, scatter to Bs[n][k]
      uint4 val = *(const uint4*)(Bb + (long)(k0 + bkr) * ldb + tn + bnc);
      const unsigned short* e = (const unsigned short*)&val;
#pragma unroll
      for (int q = 0; q < 8; ++q) Bs[bnc + q][bkr] = e[q];
    } else {       // Bt(N,K): contiguous along K -> direct b128 into Bs[n][k]
      *(uint4*)(&Bs[btn][btk]) =
          *(const uint4*)(Bb + (long)(tn + btn) * ldb + k0 + btk);
    }
    if (k0 + 32 < K) { // prefetch next K tile (global_prefetch_b8)
      __builtin_prefetch(Ab + (long)(tm + arow) * lda + k0 + 32 + acol, 0, 3);
      if (!bTrans)
        __builtin_prefetch(Bb + (long)(k0 + 32 + bkr) * ldb + tn + bnc, 0, 3);
      else
        __builtin_prefetch(Bb + (long)(tn + btn) * ldb + k0 + 32 + btk, 0, 3);
    }
    __syncthreads();

    FragBF a0, a1, b0, b1;
    int ra0 = m_off + r, ra1 = m_off + 16 + r;
    int nb0 = n_off + r, nb1 = n_off + 16 + r;
    // A fragment: K in [hi*8, hi*8+8) U [16+hi*8, 16+hi*8+8)  (2x b128)
    *(uint4*)(&a0.u32[0]) = *(const uint4*)(&As[ra0][hi * 8]);
    *(uint4*)(&a0.u32[4]) = *(const uint4*)(&As[ra0][16 + hi * 8]);
    *(uint4*)(&a1.u32[0]) = *(const uint4*)(&As[ra1][hi * 8]);
    *(uint4*)(&a1.u32[4]) = *(const uint4*)(&As[ra1][16 + hi * 8]);
    // B fragment: K in [hi*8, hi*8+8) U [16+hi*8, 16+hi*8+8) in column-major Bs (2x b128)
    *(uint4*)(&b0.u32[0]) = *(const uint4*)(&Bs[nb0][hi * 8]);
    *(uint4*)(&b0.u32[4]) = *(const uint4*)(&Bs[nb0][16 + hi * 8]);
    *(uint4*)(&b1.u32[0]) = *(const uint4*)(&Bs[nb1][hi * 8]);
    *(uint4*)(&b1.u32[4]) = *(const uint4*)(&Bs[nb1][16 + hi * 8]);

    acc00 = __builtin_amdgcn_wmma_f32_16x16x32_bf16(false, a0.v, false, b0.v,
                                                    (short)0, acc00, false, false);
    acc01 = __builtin_amdgcn_wmma_f32_16x16x32_bf16(false, a0.v, false, b1.v,
                                                    (short)0, acc01, false, false);
    acc10 = __builtin_amdgcn_wmma_f32_16x16x32_bf16(false, a1.v, false, b0.v,
                                                    (short)0, acc10, false, false);
    acc11 = __builtin_amdgcn_wmma_f32_16x16x32_bf16(false, a1.v, false, b1.v,
                                                    (short)0, acc11, false, false);
  }

  long gm_base = tm + m_off;
  int gn0 = tn + n_off + r;
  int gn1 = gn0 + 16;
  float bs0 = bias ? bias[gn0] : 0.0f;
  float bs1 = bias ? bias[gn1] : 0.0f;
#pragma unroll
  for (int j = 0; j < 8; ++j) {
    int ml = j + hi * 8;
    long gm0 = gm_base + ml;
    long gm1 = gm_base + 16 + ml;
    float v00 = acc00[j] * outScale + bs0;
    float v01 = acc01[j] * outScale + bs1;
    float v10 = acc10[j] * outScale + bs0;
    float v11 = acc11[j] * outScale + bs1;
    if (Cf) {
      float* Cfb = Cf + (long)bz * sC;
      Cfb[gm0 * ldc + gn0] = v00;
      Cfb[gm0 * ldc + gn1] = v01;
      Cfb[gm1 * ldc + gn0] = v10;
      Cfb[gm1 * ldc + gn1] = v11;
    }
    if (Cb) {
      if (headMode) {
        int h0 = gn0 >> 6, d0 = gn0 & 63;
        int h1 = gn1 >> 6, d1 = gn1 & 63;
        Cb[(((long)bz * HEADS + h0) * M + gm0) * DH + d0] = f2bf(v00);
        Cb[(((long)bz * HEADS + h1) * M + gm0) * DH + d1] = f2bf(v01);
        Cb[(((long)bz * HEADS + h0) * M + gm1) * DH + d0] = f2bf(v10);
        Cb[(((long)bz * HEADS + h1) * M + gm1) * DH + d1] = f2bf(v11);
      } else {
        unsigned short* Cbb = Cb + (long)bz * sC;
        Cbb[gm0 * ldc + gn0] = f2bf(v00);
        Cbb[gm0 * ldc + gn1] = f2bf(v01);
        Cbb[gm1 * ldc + gn0] = f2bf(v10);
        Cbb[gm1 * ldc + gn1] = f2bf(v11);
      }
    }
  }
}

// ---------------------------------------------------------------------------
// fp32 WMMA matmul (V_WMMA_F32_16X16X4_F32): C = alpha * A(M,K)*B(K,N)
// 64 threads = 2 waves, each wave one 16x16 tile (block covers 32x16)
// ---------------------------------------------------------------------------
__global__ void __launch_bounds__(64)
wmma_f32_mm_k(const float* __restrict__ A, const float* __restrict__ B,
              float* __restrict__ C, int M, int N, int K,
              int lda, int ldb, int ldc, long sA, long sB, long sC,
              float alpha) {
  int bz = blockIdx.z;
  const float* Ab = A + (long)bz * sA;
  const float* Bb = B + (long)bz * sB;
  float* Cb = C + (long)bz * sC;
  int wv = threadIdx.x >> 5, lane = threadIdx.x & 31;
  int tm = blockIdx.x * 32 + wv * 16;
  int tn = blockIdx.y * 16;
  int r = lane & 15, hi = lane >> 4;
  v8f acc = {};
  for (int k0 = 0; k0 < K; k0 += 4) {
    v2f a, b;
    const float* ap = Ab + (long)(tm + r) * lda + k0 + hi * 2;
    a[0] = ap[0];
    a[1] = ap[1];
    b[0] = Bb[(long)(k0 + hi * 2) * ldb + tn + r];
    b[1] = Bb[(long)(k0 + hi * 2 + 1) * ldb + tn + r];
    acc = __builtin_amdgcn_wmma_f32_16x16x4_f32(false, a, false, b,
                                                (short)0, acc, false, false);
  }
#pragma unroll
  for (int j = 0; j < 8; ++j) {
    int m = tm + j + hi * 8;
    Cb[(long)m * ldc + tn + r] = acc[j] * alpha;
  }
}

// ---------------------------------------------------------------------------
// Landmark means: out(bh, 256, 64) = mean over 24-row segments of x(bh,6144,64)
// ---------------------------------------------------------------------------
__global__ void landmarks_k(const unsigned short* __restrict__ x,
                            unsigned short* __restrict__ out) {
  int j = blockIdx.x, bh = blockIdx.y, d = threadIdx.x;
  const unsigned short* p = x + ((long)bh * NPAD + j * LSEG) * DH + d;
  float s = 0.0f;
  for (int t = 0; t < LSEG; ++t) s += bf2f(p[t * DH]);
  out[((long)bh * LM + j) * DH + d] = f2bf(s * (1.0f / LSEG));
}

// ---------------------------------------------------------------------------
// Row softmax: fp32 scores -> bf16 and/or fp32 output (either nullable)
// ---------------------------------------------------------------------------
__global__ void __launch_bounds__(256)
softmax_rows_k(const float* __restrict__ S, unsigned short* __restrict__ obf,
               float* __restrict__ of32, int ncols) {
  long row = blockIdx.x;
  const float* x = S + row * (long)ncols;
  __shared__ float red[256];
  int t = threadIdx.x;
  float mx = -3.4e38f;
  for (int j = t; j < ncols; j += 256) mx = fmaxf(mx, x[j]);
  red[t] = mx; __syncthreads();
  for (int s = 128; s > 0; s >>= 1) { if (t < s) red[t] = fmaxf(red[t], red[t + s]); __syncthreads(); }
  mx = red[0]; __syncthreads();
  float sum = 0.0f;
  for (int j = t; j < ncols; j += 256) sum += __expf(x[j] - mx);
  red[t] = sum; __syncthreads();
  for (int s = 128; s > 0; s >>= 1) { if (t < s) red[t] += red[t + s]; __syncthreads(); }
  float inv = 1.0f / red[0];
  for (int j = t; j < ncols; j += 256) {
    float v = __expf(x[j] - mx) * inv;
    if (obf)  obf[row * (long)ncols + j]  = f2bf(v);
    if (of32) of32[row * (long)ncols + j] = v;
  }
}

// ---------------------------------------------------------------------------
// pinv init: per-bh max |row-sum| and max |col-sum| of x(bh,256,256)
// ---------------------------------------------------------------------------
__global__ void __launch_bounds__(256)
pinv_scale_k(const float* __restrict__ x, float* __restrict__ red_out) {
  int bh = blockIdx.x, t = threadIdx.x;
  const float* m = x + (long)bh * LM * LM;
  float rs = 0.0f, cs = 0.0f;
  for (int j = 0; j < LM; ++j) {
    rs += fabsf(m[t * LM + j]);
    cs += fabsf(m[j * LM + t]);
  }
  __shared__ float r1[256], r2[256];
  r1[t] = rs; r2[t] = cs; __syncthreads();
  for (int s = 128; s > 0; s >>= 1) {
    if (t < s) { r1[t] = fmaxf(r1[t], r1[t + s]); r2[t] = fmaxf(r2[t], r2[t + s]); }
    __syncthreads();
  }
  if (t == 0) { red_out[bh] = r1[0]; red_out[BH + bh] = r2[0]; }
}

__global__ void pinv_finalize_k(float* __restrict__ red_out) {
  if (threadIdx.x == 0) {
    float mr = 0.0f, mc = 0.0f;
    for (int i = 0; i < BH; ++i) { mr = fmaxf(mr, red_out[i]); mc = fmaxf(mc, red_out[BH + i]); }
    red_out[2 * BH] = 1.0f / (mr * mc);   // global scale (matches jnp.max over all axes)
  }
}

__global__ void transpose_scale_k(const float* __restrict__ x,
                                  const float* __restrict__ invscale,
                                  float* __restrict__ z) {
  int row = blockIdx.x, bh = blockIdx.y, t = threadIdx.x;
  float s = invscale[0];
  z[((long)bh * LM + row) * LM + t] = x[((long)bh * LM + t) * LM + row] * s;
}

__global__ void id_minus_k(float alpha, const float* __restrict__ in,
                           float* __restrict__ out) {
  long i = (long)blockIdx.x * blockDim.x + threadIdx.x;
  long rc = i & (long)(LM * LM - 1);
  int rr = (int)(rc >> 8), cc = (int)(rc & (LM - 1));
  out[i] = (rr == cc ? alpha : 0.0f) - in[i];
}

// ---------------------------------------------------------------------------
// Depthwise residual conv along sequence (kernel 33, pad 16), += into out_heads
// ---------------------------------------------------------------------------
__global__ void res_conv_k(const unsigned short* __restrict__ v,
                           const float* __restrict__ w,
                           float* __restrict__ outh) {
  int i = blockIdx.x, bh = blockIdx.y, d = threadIdx.x;
  int hh = bh & (HEADS - 1);
  float s = 0.0f;
  for (int k = 0; k < RESK; ++k) {
    int src = i + k - RESK / 2;
    if (src >= 0 && src < NPAD)
      s += bf2f(v[((long)bh * NPAD + src) * DH + d]) * w[hh * RESK + k];
  }
  outh[((long)bh * NPAD + i) * DH + d] += s;
}

// ---------------------------------------------------------------------------
// (b,h,n,d) fp32 -> (b,n,h*64+d) bf16
// ---------------------------------------------------------------------------
__global__ void merge_heads_k(const float* __restrict__ oh,
                              unsigned short* __restrict__ mg) {
  long idx = (long)blockIdx.x * blockDim.x + threadIdx.x;
  long total = (long)BH * NPAD * DH;
  if (idx >= total) return;
  int d = (int)(idx & (DH - 1));
  long rem = idx / DH;
  int i = (int)(rem % NPAD);
  int bh = (int)(rem / NPAD);
  int b = bh >> 3, h = bh & (HEADS - 1);
  mg[((long)b * NPAD + i) * DMODEL + h * DH + d] = f2bf(oh[idx]);
}

// ---------------------------------------------------------------------------
// h_out = h_in + y[:, -NTOK:]   (rows 59..6143 of y)
// ---------------------------------------------------------------------------
__global__ void add_residual_k(const float* __restrict__ hin,
                               const float* __restrict__ y,
                               float* __restrict__ hout) {
  long idx = (long)blockIdx.x * blockDim.x + threadIdx.x;
  long total = (long)BATCH * NTOK * DMODEL;
  if (idx >= total) return;
  int c = (int)(idx % DMODEL);
  long row = idx / DMODEL;
  int b = (int)(row / NTOK);
  int i = (int)(row % NTOK);
  hout[idx] = hin[idx] + y[((long)b * NPAD + PADF + i) * DMODEL + c];
}

// ---------------------------------------------------------------------------
// PPEG: x + dwconv7 + dwconv5 + dwconv3 on the 78x78 map; cls row passthrough
// ---------------------------------------------------------------------------
__global__ void ppeg_k(const float* __restrict__ hin, float* __restrict__ hout,
                       const float* __restrict__ w7, const float* __restrict__ b7,
                       const float* __restrict__ w5, const float* __restrict__ b5,
                       const float* __restrict__ w3, const float* __restrict__ b3) {
  long idx = (long)blockIdx.x * blockDim.x + threadIdx.x;
  long total = (long)BATCH * NTOK * DMODEL;
  if (idx >= total) return;
  int c = (int)(idx % DMODEL);
  long row = idx / DMODEL;
  int b = (int)(row / NTOK);
  int r = (int)(row % NTOK);
  if (r == 0) { hout[idx] = hin[idx]; return; }
  int p = r - 1;
  int yy = p / HS, xx = p % HS;
  const float* feat = hin + ((long)b * NTOK + 1) * DMODEL + c;
  float acc = feat[(long)p * DMODEL] + b7[c] + b5[c] + b3[c];
  const float* W7 = w7 + (long)c * 49;
  for (int ky = 0; ky < 7; ++ky) {
    int sy = yy + ky - 3; if (sy < 0 || sy >= HS) continue;
    for (int kx = 0; kx < 7; ++kx) {
      int sx = xx + kx - 3; if (sx < 0 || sx >= HS) continue;
      acc += feat[(long)(sy * HS + sx) * DMODEL] * W7[ky * 7 + kx];
    }
  }
  const float* W5 = w5 + (long)c * 25;
  for (int ky = 0; ky < 5; ++ky) {
    int sy = yy + ky - 2; if (sy < 0 || sy >= HS) continue;
    for (int kx = 0; kx < 5; ++kx) {
      int sx = xx + kx - 2; if (sx < 0 || sx >= HS) continue;
      acc += feat[(long)(sy * HS + sx) * DMODEL] * W5[ky * 5 + kx];
    }
  }
  const float* W3 = w3 + (long)c * 9;
  for (int ky = 0; ky < 3; ++ky) {
    int sy = yy + ky - 1; if (sy < 0 || sy >= HS) continue;
    for (int kx = 0; kx < 3; ++kx) {
      int sx = xx + kx - 1; if (sx < 0 || sx >= HS) continue;
      acc += feat[(long)(sy * HS + sx) * DMODEL] * W3[ky * 3 + kx];
    }
  }
  hout[idx] = acc;
}

// ===========================================================================
extern "C" void kernel_launch(void* const* d_in, const int* in_sizes, int n_in,
                              void* d_out, int out_size, void* d_ws, size_t ws_size,
                              hipStream_t stream) {
  (void)in_sizes; (void)n_in; (void)out_size; (void)ws_size;
  const float* features = (const float*)d_in[0];
  const float* cls      = (const float*)d_in[1];
  const float* ln1_g = (const float*)d_in[2];
  const float* ln1_b = (const float*)d_in[3];
  const float* qkv1  = (const float*)d_in[4];
  const float* out1w = (const float*)d_in[5];
  const float* out1b = (const float*)d_in[6];
  const float* res1  = (const float*)d_in[7];
  const float* peg7w = (const float*)d_in[8];
  const float* peg7b = (const float*)d_in[9];
  const float* peg5w = (const float*)d_in[10];
  const float* peg5b = (const float*)d_in[11];
  const float* peg3w = (const float*)d_in[12];
  const float* peg3b = (const float*)d_in[13];
  const float* ln2_g = (const float*)d_in[14];
  const float* ln2_b = (const float*)d_in[15];
  const float* qkv2  = (const float*)d_in[16];
  const float* out2w = (const float*)d_in[17];
  const float* out2b = (const float*)d_in[18];
  const float* res2  = (const float*)d_in[19];
  const float* lnf_g = (const float*)d_in[20];
  const float* lnf_b = (const float*)d_in[21];

  char* base = (char*)d_ws;
  size_t cur = 0;
  auto alloc = [&](size_t bytes) -> void* {
    void* p = base + cur;
    cur += (bytes + 255) & ~(size_t)255;
    return p;
  };

  unsigned short* wqkv1 = (unsigned short*)alloc((size_t)DMODEL * D3 * 2);
  unsigned short* wqkv2 = (unsigned short*)alloc((size_t)DMODEL * D3 * 2);
  unsigned short* wout1 = (unsigned short*)alloc((size_t)DMODEL * DMODEL * 2);
  unsigned short* wout2 = (unsigned short*)alloc((size_t)DMODEL * DMODEL * 2);
  float* hA = (float*)alloc((size_t)BATCH * NTOK * DMODEL * 4);
  float* hB = (float*)alloc((size_t)BATCH * NTOK * DMODEL * 4);
  unsigned short* xpad = (unsigned short*)alloc((size_t)BATCH * NPAD * DMODEL * 2);
  unsigned short* qbf  = (unsigned short*)alloc((size_t)BH * NPAD * DH * 2);
  unsigned short* kbf  = (unsigned short*)alloc((size_t)BH * NPAD * DH * 2);
  unsigned short* vbf  = (unsigned short*)alloc((size_t)BH * NPAD * DH * 2);
  unsigned short* qlbf = (unsigned short*)alloc((size_t)BH * LM * DH * 2);
  unsigned short* klbf = (unsigned short*)alloc((size_t)BH * LM * DH * 2);
  float* S  = (float*)alloc((size_t)BH * NPAD * LM * 4);
  unsigned short* a1bf = (unsigned short*)alloc((size_t)BH * NPAD * LM * 2);
  unsigned short* a3bf = (unsigned short*)alloc((size_t)BH * LM * NPAD * 2);
  float* att2 = (float*)alloc((size_t)BH * LM * LM * 4);
  float* Z0 = (float*)alloc((size_t)BH * LM * LM * 4);
  float* Z1 = (float*)alloc((size_t)BH * LM * LM * 4);
  float* P1 = (float*)alloc((size_t)BH * LM * LM * 4);
  float* P2 = (float*)alloc((size_t)BH * LM * LM * 4);
  float* P3 = (float*)alloc((size_t)BH * LM * LM * 4);
  float* redbuf = (float*)alloc(64 * 4);
  float* kv  = (float*)alloc((size_t)BH * LM * DH * 4);
  float* tmp = (float*)alloc((size_t)BH * LM * DH * 4);
  unsigned short* tmpbf = (unsigned short*)alloc((size_t)BH * LM * DH * 2);
  float* oh = (float*)alloc((size_t)BH * NPAD * DH * 4);
  unsigned short* mg = (unsigned short*)alloc((size_t)BATCH * NPAD * DMODEL * 2);
  float* y = (float*)alloc((size_t)BATCH * NPAD * DMODEL * 4);

  const long totalH = (long)BATCH * NTOK * DMODEL;
  const int nbH = (int)((totalH + 255) / 256);
  const long mmsz = (long)LM * LM;

  // --- weight conversion to bf16
  {
    long n = (long)DMODEL * D3;
    convert_bf16_k<<<(int)((n + 255) / 256), 256, 0, stream>>>(qkv1, wqkv1, n);
    convert_bf16_k<<<(int)((n + 255) / 256), 256, 0, stream>>>(qkv2, wqkv2, n);
    long n2 = (long)DMODEL * DMODEL;
    convert_bf16_k<<<(int)((n2 + 255) / 256), 256, 0, stream>>>(out1w, wout1, n2);
    convert_bf16_k<<<(int)((n2 + 255) / 256), 256, 0, stream>>>(out2w, wout2, n2);
  }

  build_h_k<<<nbH, 256, 0, stream>>>(features, cls, hA);

  auto attn_block = [&](const float* hin, float* hout,
                        const unsigned short* wqkv, const unsigned short* wout,
                        const float* lng, const float* lnb,
                        const float* outb, const float* resw) {
    // LN + front pad, bf16
    ln_pad_k<<<dim3(NPAD, BATCH), 256, 0, stream>>>(hin, lng, lnb, xpad);
    // QKV GEMMs -> head-layout bf16 (q fused with SCALE)
    dim3 gq(NPAD / 128, DMODEL / 64, BATCH);
    gemm_bf16_k<<<gq, 256, 0, stream>>>(xpad, wqkv, nullptr, qbf, nullptr,
        NPAD, DMODEL, DMODEL, DMODEL, D3, 0,
        (long)NPAD * DMODEL, 0L, 0L, 0, 1, SCALEQ);
    gemm_bf16_k<<<gq, 256, 0, stream>>>(xpad, wqkv + DMODEL, nullptr, kbf, nullptr,
        NPAD, DMODEL, DMODEL, DMODEL, D3, 0,
        (long)NPAD * DMODEL, 0L, 0L, 0, 1, 1.0f);
    gemm_bf16_k<<<gq, 256, 0, stream>>>(xpad, wqkv + 2 * DMODEL, nullptr, vbf, nullptr,
        NPAD, DMODEL, DMODEL, DMODEL, D3, 0,
        (long)NPAD * DMODEL, 0L, 0L, 0, 1, 1.0f);
    // landmarks
    landmarks_k<<<dim3(LM, BH), DH, 0, stream>>>(qbf, qlbf);
    landmarks_k<<<dim3(LM, BH), DH, 0, stream>>>(kbf, klbf);
    // attn1 = softmax(q @ k_l^T)
    gemm_bf16_k<<<dim3(NPAD / 128, LM / 64, BH), 256, 0, stream>>>(qbf, klbf, S, nullptr, nullptr,
        NPAD, LM, DH, DH, DH, LM,
        (long)NPAD * DH, (long)LM * DH, (long)NPAD * LM, 1, 0, 1.0f);
    softmax_rows_k<<<BH * NPAD, 256, 0, stream>>>(S, a1bf, nullptr, LM);
    // attn2 = softmax(q_l @ k_l^T) fp32 (pinv input)
    gemm_bf16_k<<<dim3(LM / 128, LM / 64, BH), 256, 0, stream>>>(qlbf, klbf, P1, nullptr, nullptr,
        LM, LM, DH, DH, DH, LM,
        (long)LM * DH, (long)LM * DH, mmsz, 1, 0, 1.0f);
    softmax_rows_k<<<BH * LM, 256, 0, stream>>>(P1, nullptr, att2, LM);
    // attn3 = softmax(q_l @ k^T)
    gemm_bf16_k<<<dim3(LM / 128, NPAD / 64, BH), 256, 0, stream>>>(qlbf, kbf, S, nullptr, nullptr,
        LM, NPAD, DH, DH, DH, NPAD,
        (long)LM * DH, (long)NPAD * DH, (long)LM * NPAD, 1, 0, 1.0f);
    softmax_rows_k<<<BH * LM, 256, 0, stream>>>(S, a3bf, nullptr, NPAD);
    // kv = attn3 @ v
    gemm_bf16_k<<<dim3(LM / 128, DH / 64, BH), 256, 0, stream>>>(a3bf, vbf, kv, nullptr, nullptr,
        LM, DH, NPAD, NPAD, DH, DH,
        (long)LM * NPAD, (long)NPAD * DH, (long)LM * DH, 0, 0, 1.0f);
    // Moore-Penrose pinv of attn2 (fp32 WMMA)
    pinv_scale_k<<<BH, 256, 0, stream>>>(att2, redbuf);
    pinv_finalize_k<<<1, 32, 0, stream>>>(redbuf);
    transpose_scale_k<<<dim3(LM, BH), LM, 0, stream>>>(att2, redbuf + 2 * BH, Z0);
    float* zc = Z0; float* za = Z1;
    for (int it = 0; it < 6; ++it) {
      dim3 gmm(LM / 32, LM / 16, BH);
      wmma_f32_mm_k<<<gmm, 64, 0, stream>>>(att2, zc, P1, LM, LM, LM, LM, LM, LM, mmsz, mmsz, mmsz, 1.0f);
      id_minus_k<<<(int)((BH * mmsz) / 256), 256, 0, stream>>>(7.0f, P1, P2);
      wmma_f32_mm_k<<<gmm, 64, 0, stream>>>(P1, P2, P3, LM, LM, LM, LM, LM, LM, mmsz, mmsz, mmsz, 1.0f);
      id_minus_k<<<(int)((BH * mmsz) / 256), 256, 0, stream>>>(15.0f, P3, P2);
      wmma_f32_mm_k<<<gmm, 64, 0, stream>>>(P1, P2, P3, LM, LM, LM, LM, LM, LM, mmsz, mmsz, mmsz, 1.0f);
      id_minus_k<<<(int)((BH * mmsz) / 256), 256, 0, stream>>>(13.0f, P3, P2);
      wmma_f32_mm_k<<<gmm, 64, 0, stream>>>(zc, P2, za, LM, LM, LM, LM, LM, LM, mmsz, mmsz, mmsz, 0.25f);
      float* t = zc; zc = za; za = t;
    }
    // tmp = Z @ kv, then bf16
    wmma_f32_mm_k<<<dim3(LM / 32, DH / 16, BH), 64, 0, stream>>>(zc, kv, tmp,
        LM, DH, LM, LM, DH, DH, mmsz, (long)LM * DH, (long)LM * DH, 1.0f);
    convert_bf16_k<<<(int)(((long)BH * LM * DH + 255) / 256), 256, 0, stream>>>(tmp, tmpbf, (long)BH * LM * DH);
    // out_heads = attn1 @ tmp
    gemm_bf16_k<<<dim3(NPAD / 128, DH / 64, BH), 256, 0, stream>>>(a1bf, tmpbf, oh, nullptr, nullptr,
        NPAD, DH, LM, LM, DH, DH,
        (long)NPAD * LM, (long)LM * DH, (long)NPAD * DH, 0, 0, 1.0f);
    // += depthwise residual conv on v
    res_conv_k<<<dim3(NPAD, BH), DH, 0, stream>>>(vbf, resw, oh);
    // merge heads, project, residual add with [:, -NTOK:] slice
    merge_heads_k<<<(int)(((long)BH * NPAD * DH) / 256), 256, 0, stream>>>(oh, mg);
    gemm_bf16_k<<<dim3(BATCH * NPAD / 128, DMODEL / 64, 1), 256, 0, stream>>>(mg, wout, y, nullptr, outb,
        BATCH * NPAD, DMODEL, DMODEL, DMODEL, DMODEL, DMODEL,
        0L, 0L, 0L, 0, 0, 1.0f);
    add_residual_k<<<nbH, 256, 0, stream>>>(hin, y, hout);
  };

  attn_block(hA, hB, wqkv1, wout1, ln1_g, ln1_b, out1b, res1);
  ppeg_k<<<nbH, 256, 0, stream>>>(hB, hA, peg7w, peg7b, peg5w, peg5b, peg3w, peg3b);
  attn_block(hA, hB, wqkv2, wout2, ln2_g, ln2_b, out2b, res2);
  ln_out_k<<<dim3(NTOK, BATCH), 256, 0, stream>>>(hB, lnf_g, lnf_b, (float*)d_out);
}
